// AdaptiveSoftmaxAccWrapper_42133629174006
// MI455X (gfx1250) — matrix operand order repaired and checked by reading, not compiled, not run on hardware
//
#include <hip/hip_runtime.h>
#include <stdint.h>

// Problem constants (fixed by the reference's setup_inputs()).
#define VOCAB  50257
#define DMODEL 512
#define NROWS  8192   // 4 * 2048
#define TOPK   10

typedef __attribute__((ext_vector_type(16))) __bf16 bf16x16;
typedef __attribute__((ext_vector_type(8)))  float  v8f;
typedef __attribute__((ext_vector_type(4)))  uint32_t u32x4;
typedef __attribute__((ext_vector_type(8)))  int      i32x8;
typedef __attribute__((ext_vector_type(4)))  int      i32x4;

// ---------------------------------------------------------------------------
// Kernel A: exact f32 threshold per row: thr[n] = dot(x[n,:], W[:,t]) + b[t].
// Also zero-initializes the per-row exceed counters. One wave32 per row.
// ---------------------------------------------------------------------------
__global__ __launch_bounds__(256) void thr_kernel(
    const float* __restrict__ x, const int* __restrict__ tgt,
    const float* __restrict__ W, const float* __restrict__ b,
    float* __restrict__ thr, int* __restrict__ cnt) {
  const int wave = threadIdx.x >> 5;
  const int lane = threadIdx.x & 31;
  const int row  = blockIdx.x * 8 + wave;
  if (row >= NROWS) return;
  const int t = tgt[row];
  const float* xr = x + (size_t)row * DMODEL;
  float s = 0.f;
  #pragma unroll 4
  for (int k = lane; k < DMODEL; k += 32)
    s += xr[k] * W[(size_t)k * VOCAB + t];
  #pragma unroll
  for (int off = 16; off > 0; off >>= 1)
    s += __shfl_xor(s, off, 32);
  if (lane == 0) {
    thr[row] = s + b[t];
    cnt[row] = 0;
  }
}

// ---------------------------------------------------------------------------
// Kernel A2: one-time f32 -> bf16 conversion of x into a pre-swizzled image.
// Per row: 1024B = two 512B K-chunk blocks (K 0..255, 256..511); within a
// block, 16B chunks stored at (kc16 ^ (row%16)) so a TDM row-copy into LDS
// lands exactly in the layout the WMMA A-fragment b128 reads expect.
// ---------------------------------------------------------------------------
__global__ __launch_bounds__(256) void convert_x_kernel(
    const float* __restrict__ x, unsigned char* __restrict__ img) {
  const int q = blockIdx.x * 256 + threadIdx.x;   // quad id 0..1048575
  const int R = q >> 7;                           // row 0..8191
  const int k = (q & 127) << 2;                   // 0..508
  const float4 f = *(const float4*)(x + (size_t)R * DMODEL + k);
  union { __bf16 e[4]; uint64_t u; } p;
  p.e[0] = (__bf16)f.x; p.e[1] = (__bf16)f.y;
  p.e[2] = (__bf16)f.z; p.e[3] = (__bf16)f.w;
  const int kchunk = k >> 8;                      // 0/1
  const int kc16   = (k & 255) >> 3;              // 0..31
  const int swz    = kc16 ^ (R & 15);
  *(uint64_t*)(img + (size_t)R * 1024 + kchunk * 512 + swz * 16 + (k & 7) * 2)
      = p.u;
}

// ---------------------------------------------------------------------------
// Kernel B: bf16 WMMA GEMM that never materializes z. Counts, per row, how
// many vocab columns (excluding the target itself) have z > thr[row].
//
// Block: 256 threads (8 wave32). Tile: M=128 x N=128. K=512 as 2 chunks of
// 256; each chunk = 64KB bf16 in LDS. USE_TDM: chunk staged by a single
// tensor_load_to_lds (2D D#: tile 512B x 128 rows, row stride 1024B) from
// the pre-swizzled image; else converted in-kernel from f32 x.
// ---------------------------------------------------------------------------
template <bool USE_TDM>
__global__ __launch_bounds__(256) void gemm_count_kernel(
    const float* __restrict__ x, const unsigned char* __restrict__ img,
    const int* __restrict__ tgt,
    const float* __restrict__ W, const float* __restrict__ b,
    const float* __restrict__ thr, int* __restrict__ cnt) {
  __shared__ __align__(16) unsigned char Albs[64 * 1024];

  const int tid  = threadIdx.x;
  const int wave = tid >> 5;
  const int lane = tid & 31;
  const int r    = lane & 15;        // N within 16-wide tile / row within frag
  const int half = lane >> 4;        // 0: lanes 0-15, 1: lanes 16-31
  const int mB   = blockIdx.y * 128;
  const int n0   = blockIdx.x * 128 + wave * 16;
  const int col  = n0 + r;
  const int coln = (col < VOCAB) ? col : (VOCAB - 1);  // clamp loads

  v8f acc[8];
  #pragma unroll
  for (int m = 0; m < 8; ++m) acc[m] = 0.f;

  for (int chunk = 0; chunk < 2; ++chunk) {
    const int kbase = chunk * 256;
    if (chunk) __syncthreads();   // prior chunk's LDS reads complete

    if constexpr (USE_TDM) {
      // ---- Stage A chunk with the Tensor Data Mover (wave 0 issues) ----
      if (wave == 0) {
        const uint32_t lds_off = (uint32_t)(uintptr_t)(&Albs[0]);
        const uint64_t ga = (uint64_t)(uintptr_t)(img + (size_t)mB * 1024
                                                  + (size_t)chunk * 512);
        // D# group0: count=1, lds_addr, 57-bit global_addr, type=2.
        u32x4 g0;
        g0.x = 1u;
        g0.y = lds_off;
        g0.z = (uint32_t)ga;
        g0.w = (uint32_t)((ga >> 32) & 0x1FFFFFFu) | (2u << 30);
        // D# group1: data_size=2 (4B units); tensor_dim0=128, tensor_dim1=128,
        // tile_dim0=128 (=512B), tile_dim1=128 rows, dim0 stride=256 (=1024B).
        i32x8 g1;
        g1.s0 = (int)(2u << 16);          // wg_mask=0, data_size=4B
        g1.s1 = (int)(128u << 16);        // abar=0 | tensor_dim0[15:0]<<16
        g1.s2 = (int)(128u << 16);        // tensor_dim0[31:16]=0 | tensor_dim1[15:0]<<16
        g1.s3 = (int)(128u << 16);        // tensor_dim1[31:16]=0 | tile_dim0<<16
        g1.s4 = (int)128u;                // tile_dim1=128, tile_dim2=0
        g1.s5 = (int)256u;                // tensor_dim0_stride[31:0] (4B units)
        g1.s6 = 0;                        // stride hi, dim1_stride lo
        g1.s7 = 0;
        const i32x4 gz4 = {0, 0, 0, 0};          // groups 2/3 unused (2D)
        const i32x8 gz8 = {0, 0, 0, 0, 0, 0, 0, 0};
        __builtin_amdgcn_tensor_load_to_lds(g0, g1, gz4, gz4, gz8, 0);
        __builtin_amdgcn_s_wait_tensorcnt(0);
      }
      __syncthreads();
    } else {
      // ---- Fallback: stage A chunk f32 -> bf16, swizzled 16B chunks ----
      #pragma unroll 4
      for (int i = 0; i < 32; ++i) {
        const int c    = tid + 256 * i;        // 0..8191 quad-id
        const int rowl = c >> 6;               // 0..127
        const int k    = (c & 63) << 2;        // 0..252 within chunk
        const float4 f =
            *(const float4*)(x + (size_t)(mB + rowl) * DMODEL + kbase + k);
        union { __bf16 e[4]; uint64_t u; } p;
        p.e[0] = (__bf16)f.x; p.e[1] = (__bf16)f.y;
        p.e[2] = (__bf16)f.z; p.e[3] = (__bf16)f.w;
        const int kc16 = k >> 3;
        const int swz  = kc16 ^ (rowl & 15);
        *(uint64_t*)(Albs + rowl * 512 + swz * 16 + (k & 7) * 2) = p.u;
      }
      __syncthreads();
    }

    for (int ks = 0; ks < 8; ++ks) {
      const int k0 = kbase + ks * 32;
      if (k0 + 32 < DMODEL) {  // hint next W K-slab (global_prefetch_b8)
        __builtin_prefetch(W + (size_t)(k0 + 32 + half * 16) * VOCAB + coln, 0, 1);
        __builtin_prefetch(W + (size_t)(k0 + 40 + half * 16) * VOCAB + coln, 0, 1);
      }

      // B fragment: 32x16 bf16. N = lane%16; lanes<16 hold K=0..7,16..23
      // (2 per VGPR), lanes>=16 hold K=8..15,24..31.
      union { bf16x16 v; __bf16 e[16]; } bf;
      #pragma unroll
      for (int s = 0; s < 8; ++s) {
        const int ka = k0 + half * 8 + 2 * s + ((s >= 4) ? 8 : 0);
        const float f0 = W[(size_t)ka       * VOCAB + coln];
        const float f1 = W[(size_t)(ka + 1) * VOCAB + coln];
        bf.e[2 * s]     = (__bf16)f0;
        bf.e[2 * s + 1] = (__bf16)f1;
      }

      #pragma unroll
      for (int m = 0; m < 8; ++m) {
        const int lr = m * 16 + r;  // local A row (rows striped across lanes)
        union { bf16x16 v; uint4 u[2]; } af;
        const int kcA = ks * 4 + half;       // run0: K = ks*32 + half*8
        const int kcB = ks * 4 + 2 + half;   // run1: +16
        af.u[0] = *(const uint4*)(Albs + lr * 512 + ((kcA ^ r) * 16));
        af.u[1] = *(const uint4*)(Albs + lr * 512 + ((kcB ^ r) * 16));
        acc[m] = __builtin_amdgcn_wmma_f32_16x16x32_bf16(
            false, af.v, false, bf.v, (short)0, acc[m], false, false);
      }
    }
  }

  // ---- Epilogue: count z > thr per row. Reuse A-LDS as counters. ----
  __syncthreads();                       // all A reads complete
  int* cbuf = (int*)Albs;
  if (tid < 128) cbuf[tid] = 0;
  __syncthreads();

  const bool  cvalid = (col < VOCAB);
  const float bias   = cvalid ? b[col] : 0.f;
  #pragma unroll
  for (int m = 0; m < 8; ++m) {
    #pragma unroll
    for (int v = 0; v < 8; ++v) {
      // C/D layout: VGPR v <-> M = v (lanes 0-15) / v+8 (lanes 16-31); N = r.
      const int row = mB + m * 16 + v + 8 * half;
      const float z = acc[m][v] + bias;
      const bool pred = cvalid && (z > thr[row]) && (col != tgt[row]);
      const unsigned int msk = (unsigned int)__ballot(pred);
      if (lane == 0) {
        const int clo = __popc(msk & 0xFFFFu);
        const int chi = __popc(msk >> 16);
        if (clo) atomicAdd(&cbuf[m * 16 + v],     clo);
        if (chi) atomicAdd(&cbuf[m * 16 + v + 8], chi);
      }
    }
  }
  __syncthreads();
  if (tid < 128) {
    const int c = cbuf[tid];
    if (c) atomicAdd(&cnt[mB + tid], c);
  }
}

// ---------------------------------------------------------------------------
// Kernel C: hit = (cnt < K); out = sum(hit*mask) / sum(mask). Single block.
// ---------------------------------------------------------------------------
__global__ __launch_bounds__(256) void finalize_kernel(
    const int* __restrict__ cnt, const int* __restrict__ mask,
    float* __restrict__ out) {
  __shared__ int s_hit, s_tot;
  if (threadIdx.x == 0) { s_hit = 0; s_tot = 0; }
  __syncthreads();
  int h = 0, t = 0;
  for (int i = threadIdx.x; i < NROWS; i += 256) {
    const int m = mask[i];
    t += m;
    if (m && cnt[i] < TOPK) ++h;
  }
  #pragma unroll
  for (int off = 16; off > 0; off >>= 1) {
    h += __shfl_xor(h, off, 32);
    t += __shfl_xor(t, off, 32);
  }
  if ((threadIdx.x & 31) == 0) { atomicAdd(&s_hit, h); atomicAdd(&s_tot, t); }
  __syncthreads();
  if (threadIdx.x == 0) out[0] = (float)s_hit / (float)s_tot;
}

// ---------------------------------------------------------------------------
extern "C" void kernel_launch(void* const* d_in, const int* in_sizes, int n_in,
                              void* d_out, int out_size, void* d_ws, size_t ws_size,
                              hipStream_t stream) {
  const float* x    = (const float*)d_in[0];   // [4,2048,512] f32
  const int*   tgt  = (const int*)d_in[1];     // [4,2048] i32
  const int*   mask = (const int*)d_in[2];     // [4,2048] i32
  const float* W    = (const float*)d_in[3];   // [512,50257] f32
  const float* b    = (const float*)d_in[4];   // [50257] f32
  float* out = (float*)d_out;

  float* thr = (float*)d_ws;                                   // 8192 f32
  int*   cnt = (int*)((char*)d_ws + NROWS * sizeof(float));    // 8192 i32
  unsigned char* img = (unsigned char*)d_ws + 64 * 1024;       // 8MB bf16 image
  const size_t need = 64 * 1024 + (size_t)NROWS * DMODEL * 2;

  thr_kernel<<<NROWS / 8, 256, 0, stream>>>(x, tgt, W, b, thr, cnt);

  dim3 grid((VOCAB + 127) / 128, NROWS / 128);   // (393, 64)
  if (ws_size >= need) {
    convert_x_kernel<<<(NROWS * DMODEL / 4) / 256, 256, 0, stream>>>(x, img);
    gemm_count_kernel<true><<<grid, 256, 0, stream>>>(x, img, tgt, W, b, thr, cnt);
  } else {
    gemm_count_kernel<false><<<grid, 256, 0, stream>>>(x, nullptr, tgt, W, b, thr, cnt);
  }

  finalize_kernel<<<1, 256, 0, stream>>>(cnt, mask, out);
}
